// DynamicRouting_10453950399034
// MI455X (gfx1250) — matrix operand order repaired and checked by reading, not compiled, not run on hardware
//
#include <hip/hip_runtime.h>

// ---------------------------------------------------------------------------
// DynamicRouting fused kernel for gfx1250 (MI455X).
//   con[g,o,p] via v_wmma_f32_16x16x32_bf16, routing in-register,
//   beta cross-wave reduction via LDS ds_add_f32, weights staged once
//   per workgroup into 256KB of the 320KB WGP LDS in WMMA A-fragment order.
//   x loads / out stores are non-temporal (stream-once data) so the
//   L2-resident weight tensor is not evicted.
// ---------------------------------------------------------------------------

typedef __attribute__((ext_vector_type(16))) __bf16 v16bf;
typedef __attribute__((ext_vector_type(8)))  float  v8f;
typedef __attribute__((ext_vector_type(4)))  float  v4f;   // native vec for NT loads

#define NGROUP   8
#define FI       64
#define FO       256
#define CIN      512
#define BATCH    16
#define HW       1024
#define NPOS     (BATCH * HW)     // 16384 spatial positions
#define TILEP    16               // positions per workgroup tile (WMMA N)
#define NTILES   (NPOS / TILEP)   // 1024
#define NTHREADS 256              // 8 wave32 waves (2/SIMD -> VGPR-file safe)
#define NWG      256              // covers up to 256 WGPs; 4 tiles per WG

#define WFRAG_BYTES (256 * 1024)  // 8g * 16 otile * 2 kchunk frags, 1KB each
#define XFRAG_BYTES (16 * 1024)   // 8g * 2 kchunk frags, 1KB each
#define BETA_OFF    (WFRAG_BYTES + XFRAG_BYTES)
#define SMEM_BYTES  (BETA_OFF + NGROUP * TILEP * 4)   // 279,040 B < 320KB WGP LDS

// float -> bf16, round-to-nearest-even
__device__ __forceinline__ unsigned int f2bf(float f) {
    unsigned int u = __float_as_uint(f);
    u += 0x7FFFu + ((u >> 16) & 1u);
    return u >> 16;
}

__device__ __forceinline__ float sigmoidf_fast(float x) {
    return 1.0f / (1.0f + __expf(-x));
}

__global__ __launch_bounds__(NTHREADS, 1)
void dynroute_kernel(const float* __restrict__ x,
                     const float* __restrict__ w,
                     const float* __restrict__ bias,
                     float* __restrict__ out)
{
    extern __shared__ unsigned char smem[];
    unsigned char* wlds    = smem;                     // weight fragments (bf16)
    unsigned char* xlds    = smem + WFRAG_BYTES;       // x fragments (bf16)
    float*         betalds = (float*)(smem + BETA_OFF);

    const int tid  = threadIdx.x;
    const int wave = tid >> 5;
    const int lane = tid & 31;
    const int lrow = lane & 15;   // WMMA N column == position-in-tile
    const int lhi  = lane >> 4;   // C-matrix M offset selector (+8 rows)

    // ---- Stage weights once per WG into LDS, WMMA A-fragment order ----
    // frag id fid = ((g*16 + otile)*2 + kchunk); lane slot = 32B = 16 bf16.
    // A k order used here: lane half kb=16*(l>=16); elem j -> K=kb+j
    //                      (matches B fragment order).
    for (int s = tid; s < 256 * 32; s += NTHREADS) {
        const int fid = s >> 5;
        const int l   = s & 31;
        const int kc  = fid & 1;
        const int ot  = (fid >> 1) & 15;
        const int g   = fid >> 5;
        const int o   = ot * 16 + (l & 15);
        const int kb  = (l < 16) ? 0 : 16;
        const float* wr = w + (size_t)o * CIN + g * FI + kc * 32 + kb;
        unsigned int pk[8];
#pragma unroll
        for (int q = 0; q < 4; ++q)
            pk[q]     = f2bf(wr[2 * q])      | (f2bf(wr[2 * q + 1])      << 16);
#pragma unroll
        for (int q = 0; q < 4; ++q)
            pk[4 + q] = f2bf(wr[8 + 2 * q]) | (f2bf(wr[8 + 2 * q + 1]) << 16);
        uint4* dst = (uint4*)(wlds + (size_t)fid * 1024 + l * 32);
        dst[0] = make_uint4(pk[0], pk[1], pk[2], pk[3]);
        dst[1] = make_uint4(pk[4], pk[5], pk[6], pk[7]);
    }

    // ---- Preload this lane's bias values (C layout: o = otile*16 + r + 8*lhi)
    float bi[2][8];
#pragma unroll
    for (int ot2 = 0; ot2 < 2; ++ot2)
#pragma unroll
        for (int r = 0; r < 8; ++r)
            bi[ot2][r] = bias[(wave * 2 + ot2) * 16 + r + 8 * lhi];

    // ---- Tile loop: 16 positions per iteration, grid-strided ----
    for (int tile = blockIdx.x; tile < NTILES; tile += gridDim.x) {
        const int    pos0 = tile * TILEP;         // never crosses batch bndry
        const int    b    = pos0 >> 10;
        const int    hw0  = pos0 & (HW - 1);
        const float* xb   = x + (size_t)b * CIN * HW + hw0;

        if (tid < NGROUP * TILEP) betalds[tid] = 0.0f;

        // Stage x tile into B-fragment order: thread owns rows c=2t, 2t+1.
        // B k order used here: lanes 0-15 hold K=0..15 (elem j=K), lanes 16-31 K=16..31.
        // Non-temporal: x is streamed exactly once.
        {
            const int c0 = tid * 2;
            const int g  = c0 >> 6;
            const int kk = c0 & 63;
            const int kc = kk >> 5;
            const int K  = kk & 31;               // even
            unsigned char* fb = xlds + (size_t)(g * 2 + kc) * 1024 + (K & 15) * 2;
            const v4f* r0 = (const v4f*)(xb + (size_t)c0 * HW);
            const v4f* r1 = (const v4f*)(xb + (size_t)(c0 + 1) * HW);
            v4f a[4], c[4];
#pragma unroll
            for (int q = 0; q < 4; ++q) {
                a[q] = __builtin_nontemporal_load(r0 + q);
                c[q] = __builtin_nontemporal_load(r1 + q);
            }
            const float* af = (const float*)a;
            const float* cf = (const float*)c;
#pragma unroll
            for (int p = 0; p < 16; ++p) {
                const unsigned int pk = f2bf(af[p]) | (f2bf(cf[p]) << 16);
                const int lslot = p + ((K >= 16) ? 16 : 0);
                *(unsigned int*)(fb + lslot * 32) = pk;
            }
            // Prefetch next tile's rows (global_prefetch_b8, speculative)
            const int nt = tile + gridDim.x;
            if (nt < NTILES) {
                const int np = nt * TILEP;
                __builtin_prefetch(x + (size_t)(np >> 10) * CIN * HW +
                                       (np & (HW - 1)) + (size_t)c0 * HW, 0, 1);
            }
        }
        __syncthreads();

        // ---- con[g, o(32 per wave), p] via 32 WMMAs ----
        v8f acc[NGROUP][2];
#pragma unroll
        for (int g = 0; g < NGROUP; ++g) {
#pragma unroll
            for (int ot2 = 0; ot2 < 2; ++ot2) {
                const int ot = wave * 2 + ot2;
                v8f c = {};
#pragma unroll
                for (int kc = 0; kc < 2; ++kc) {
                    const v16bf* ap = (const v16bf*)(wlds +
                        (size_t)((g * 16 + ot) * 2 + kc) * 1024 + lane * 32);
                    const v16bf* bp = (const v16bf*)(xlds +
                        (size_t)(g * 2 + kc) * 1024 + lane * 32);
                    c = __builtin_amdgcn_wmma_f32_16x16x32_bf16(
                            false, *ap, false, *bp, (short)0, c, false, false);
                }
                acc[g][ot2] = c;
            }
        }

        // ---- Routing iteration 0: alpha == sigmoid(0) == 0.5 (scalar) ----
        float v[2][8];
#pragma unroll
        for (int ot2 = 0; ot2 < 2; ++ot2)
#pragma unroll
            for (int r = 0; r < 8; ++r) {
                float s = 0.0f;
#pragma unroll
                for (int g = 0; g < NGROUP; ++g) s += acc[g][ot2][r];
                v[ot2][r] = 0.5f * s;
            }
#pragma unroll
        for (int g = 0; g < NGROUP; ++g) {
            float s = 0.0f;
#pragma unroll
            for (int ot2 = 0; ot2 < 2; ++ot2)
#pragma unroll
                for (int r = 0; r < 8; ++r) s += v[ot2][r] * acc[g][ot2][r];
            atomicAdd(&betalds[g * TILEP + lrow], s);   // ds_add_f32
        }
        __syncthreads();

        // ---- Routing iteration 1 ----
        float alpha[NGROUP];
#pragma unroll
        for (int g = 0; g < NGROUP; ++g)
            alpha[g] = sigmoidf_fast(betalds[g * TILEP + lrow]);
#pragma unroll
        for (int ot2 = 0; ot2 < 2; ++ot2)
#pragma unroll
            for (int r = 0; r < 8; ++r) {
                float s = 0.0f;
#pragma unroll
                for (int g = 0; g < NGROUP; ++g) s += alpha[g] * acc[g][ot2][r];
                v[ot2][r] = s;
            }
#pragma unroll
        for (int g = 0; g < NGROUP; ++g) {
            float s = 0.0f;
#pragma unroll
            for (int ot2 = 0; ot2 < 2; ++ot2)
#pragma unroll
                for (int r = 0; r < 8; ++r) s += v[ot2][r] * acc[g][ot2][r];
            atomicAdd(&betalds[g * TILEP + lrow], s);
        }
        __syncthreads();

        // ---- Routing iteration 2 (final): v + bias -> out (non-temporal) ----
#pragma unroll
        for (int g = 0; g < NGROUP; ++g)
            alpha[g] = sigmoidf_fast(betalds[g * TILEP + lrow]);
#pragma unroll
        for (int ot2 = 0; ot2 < 2; ++ot2)
#pragma unroll
            for (int r = 0; r < 8; ++r) {
                float s = 0.0f;
#pragma unroll
                for (int g = 0; g < NGROUP; ++g) s += alpha[g] * acc[g][ot2][r];
                const int o = (wave * 2 + ot2) * 16 + r + 8 * lhi;
                __builtin_nontemporal_store(
                    s + bi[ot2][r],
                    out + ((size_t)b * FO + o) * HW + hw0 + lrow);
            }
        __syncthreads();   // protect betalds/xlds reuse next tile
    }
}

extern "C" void kernel_launch(void* const* d_in, const int* in_sizes, int n_in,
                              void* d_out, int out_size, void* d_ws, size_t ws_size,
                              hipStream_t stream) {
    const float* x    = (const float*)d_in[0];   // [16, 512, 32, 32]
    const float* wght = (const float*)d_in[1];   // [256, 512, 1, 1]
    const float* bias = (const float*)d_in[2];   // [1, 256, 1, 1]
    float*       out  = (float*)d_out;           // [16, 256, 32, 32]
    (void)in_sizes; (void)n_in; (void)out_size; (void)d_ws; (void)ws_size;

    dynroute_kernel<<<NWG, NTHREADS, SMEM_BYTES, stream>>>(x, wght, bias, out);
}
